// FastSpeech2_24489903522674
// MI455X (gfx1250) — matrix operand order repaired and checked by reading, not compiled, not run on hardware
//
#include <hip/hip_runtime.h>
#include <cmath>

// ---------------------------------------------------------------------------
// FastSpeech2 forward on gfx1250 (MI455X).  All GEMMs run through
// v_wmma_f32_16x16x32_bf16 (bf16 operands, f32 accum).  The GEMM K-loop is
// double-buffered: tile k+1 is prefetched into registers while WMMAs consume
// tile k from LDS (one barrier per iteration).  Tile fills are branch-free:
// K dims are padded to multiples of 32 with zeroed operands; M/N ragged edges
// use index-clamped vector loads (out-of-range results are never stored).
// Workspace requirement: ~365 MB.
// ---------------------------------------------------------------------------

#define DEV __device__ __forceinline__

typedef __attribute__((ext_vector_type(16))) __bf16        v16bf;
typedef __attribute__((ext_vector_type(8)))  float         v8f;
typedef __attribute__((ext_vector_type(4)))  unsigned int  u32x4;

union FRAG { v16bf v; u32x4 q[2]; };
union VU8  { u32x4 q; unsigned short s[8]; };

DEV unsigned short f2bf(float f) {
    union { float f; unsigned u; } v; v.f = f;
    unsigned r = v.u + 0x7FFFu + ((v.u >> 16) & 1u);   // round-to-nearest-even
    return (unsigned short)(r >> 16);
}

DEV float posenc(int t, int d, int Dd) {
    int i2 = d & ~1;
    float freq = __expf(-(float)i2 * (9.210340371976184f / (float)Dd)); // ln(1e4)
    float ang = (float)t * freq;
    return (d & 1) ? __cosf(ang) : __sinf(ang);
}

// ---- fragment loaders: two aligned 16B LDS loads each ----------------------
DEV v16bf load_frag_a(const unsigned short* As, int lane) {   // row-major [m][k]
    int m = lane & 15, koff = (lane >> 4) << 3;               // 0 or 8
    const unsigned short* p = As + m * 32 + koff;
    FRAG f;
    f.q[0] = *(const u32x4*)(p);        // VGPR0-3: k = koff..koff+7
    f.q[1] = *(const u32x4*)(p + 16);   // VGPR4-7: k = 16+koff..16+koff+7
    return f.v;
}
DEV v16bf load_frag_b(const unsigned short* Bs, int lane) {   // col-major [n][k]
    int n = lane & 15, koff = (lane >> 4) << 3;               // 0 or 8
    const unsigned short* p = Bs + n * 32 + koff;
    FRAG f;
    f.q[0] = *(const u32x4*)(p);
    f.q[1] = *(const u32x4*)(p + 16);
    return f.v;
}

// ---------------- WMMA GEMM: C = act(scale*A@B + bias) [+resid] ------------
// 128 threads = 4 waves (2x2), block tile 64x64, each wave 32x32 via 4 WMMAs.
// Double-buffered LDS; prefetch of tile k+1 overlaps WMMAs on tile k.
// REQUIRES: K % 32 == 0; N % 8 == 0; operand rows beyond M / cols beyond N
// readable (index-clamped); K-padding regions of A and B must be zero.
__launch_bounds__(128)
__global__ void k_gemm(const unsigned short* __restrict__ A, long sAz, int lda,
                       const unsigned short* __restrict__ Bm, long sBz, int ldb, int transB,
                       const float* __restrict__ bias,
                       const float* __restrict__ resid,
                       float* outF, long sCFz, int ldcF,
                       unsigned short* outB, long sCBz, int ldcB,
                       int M, int N, int K, float scale, int relu,
                       int transT, int Tout) {
    __shared__ unsigned short As[2][64 * 32];   // row-major  [m][k]
    __shared__ unsigned short Bs[2][64 * 32];   // col-major  [n][k]

    int z = blockIdx.z;
    A  += (long)z * sAz;
    Bm += (long)z * sBz;
    if (outF) outF += (long)z * sCFz;
    if (outB) outB += (long)z * sCBz;

    int m0 = blockIdx.x * 64, n0 = blockIdx.y * 64;
    int tid = threadIdx.x, wave = tid >> 5, lane = tid & 31;
    int wm = (wave >> 1) * 32, wn = (wave & 1) * 32;

    // per-thread chunk coordinates (branch-free, index-clamped)
    int ar[2], akc[2]; long aoff[2];
    int bn[2], bkc[2]; long boff[2];          // transB path
    int br2[2], bnc[2]; long boff2[2];        // non-trans path
#pragma unroll
    for (int c = 0; c < 2; ++c) {
        int ch = tid + c * 128;
        ar[c] = ch >> 2; akc[c] = (ch & 3) << 3;
        int gm = m0 + ar[c]; gm = gm < M ? gm : M - 1;
        aoff[c] = (long)gm * lda + akc[c];
        bn[c] = ch >> 2; bkc[c] = (ch & 3) << 3;
        int gn = n0 + bn[c]; gn = gn < N ? gn : N - 1;
        boff[c] = (long)gn * ldb + bkc[c];
        br2[c] = ch >> 3; bnc[c] = (ch & 7) << 3;
        int gn2 = n0 + bnc[c]; gn2 = gn2 <= N - 8 ? gn2 : N - 8;
        boff2[c] = (long)br2[c] * ldb + gn2;
    }

    v8f acc[2][2];
#pragma unroll
    for (int i = 0; i < 2; ++i)
#pragma unroll
        for (int j = 0; j < 2; ++j) acc[i][j] = {};

    const int nK = K >> 5;                      // K % 32 == 0
    u32x4 ra[2], rb[2];

    // prologue: fetch tile 0 and stage into buffer 0
#pragma unroll
    for (int c = 0; c < 2; ++c) {
        ra[c] = *(const u32x4*)(A + aoff[c]);
        rb[c] = transB ? *(const u32x4*)(Bm + boff[c])
                       : *(const u32x4*)(Bm + boff2[c]);
    }
#pragma unroll
    for (int c = 0; c < 2; ++c) {
        *(u32x4*)(As[0] + ar[c] * 32 + akc[c]) = ra[c];
        if (transB) {
            *(u32x4*)(Bs[0] + bn[c] * 32 + bkc[c]) = rb[c];
        } else {
            VU8 vv; vv.q = rb[c];
#pragma unroll
            for (int j = 0; j < 8; ++j) Bs[0][(bnc[c] + j) * 32 + br2[c]] = vv.s[j];
        }
    }
    __syncthreads();

    for (int kt = 0; kt < nK; ++kt) {
        int cur = kt & 1;
        long knx = (long)(((kt + 1) < nK ? (kt + 1) : kt) << 5);  // clamped prefetch
        // issue next-tile global loads (overlap with WMMAs below)
#pragma unroll
        for (int c = 0; c < 2; ++c) {
            ra[c] = *(const u32x4*)(A + aoff[c] + knx);
            rb[c] = transB ? *(const u32x4*)(Bm + boff[c] + knx)
                           : *(const u32x4*)(Bm + boff2[c] + knx * ldb);
        }

        v16bf afr[2], bfr[2];
        afr[0] = load_frag_a(As[cur] + (wm + 0)  * 32, lane);
        afr[1] = load_frag_a(As[cur] + (wm + 16) * 32, lane);
        bfr[0] = load_frag_b(Bs[cur] + (wn + 0)  * 32, lane);
        bfr[1] = load_frag_b(Bs[cur] + (wn + 16) * 32, lane);
#pragma unroll
        for (int mi = 0; mi < 2; ++mi)
#pragma unroll
            for (int ni = 0; ni < 2; ++ni)
                acc[mi][ni] = __builtin_amdgcn_wmma_f32_16x16x32_bf16(
                    false, afr[mi], false, bfr[ni], (short)0, acc[mi][ni], false, false);

        // stage next tile into the other buffer
#pragma unroll
        for (int c = 0; c < 2; ++c) {
            *(u32x4*)(As[1 - cur] + ar[c] * 32 + akc[c]) = ra[c];
            if (transB) {
                *(u32x4*)(Bs[1 - cur] + bn[c] * 32 + bkc[c]) = rb[c];
            } else {
                VU8 vv; vv.q = rb[c];
#pragma unroll
                for (int j = 0; j < 8; ++j)
                    Bs[1 - cur][(bnc[c] + j) * 32 + br2[c]] = vv.s[j];
            }
        }
        __syncthreads();
    }

    int mlo = (lane >> 4) << 3;                            // +0 / +8
    int nl  = lane & 15;
#pragma unroll
    for (int mi = 0; mi < 2; ++mi)
#pragma unroll
        for (int ni = 0; ni < 2; ++ni)
#pragma unroll
            for (int i = 0; i < 8; ++i) {
                int m = m0 + wm + mi * 16 + i + mlo;
                int n = n0 + wn + ni * 16 + nl;
                if (m < M && n < N) {
                    float v = acc[mi][ni][i] * scale;
                    if (bias)  v += bias[n];
                    if (relu)  v = fmaxf(v, 0.f);
                    if (resid) v += resid[(long)m * ldcF + n];
                    if (outF) {
                        long idx;
                        if (transT) {
                            int bb = m / Tout, tt = m - bb * Tout;
                            idx = ((long)bb * N + n) * (long)Tout + tt;
                        } else idx = (long)m * ldcF + n;
                        outF[idx] = v;
                    }
                    if (outB) outB[(long)m * ldcB + n] = f2bf(v);
                }
            }
}

// ---------------- LayerNorm: one wave (32 lanes) per row --------------------
__launch_bounds__(256)
__global__ void k_ln(const float* __restrict__ x, const float* __restrict__ g,
                     const float* __restrict__ b, unsigned short* outB, float* outF,
                     int M, int Dd) {
    int row = blockIdx.x * 8 + (threadIdx.x >> 5);
    int lane = threadIdx.x & 31;
    if (row >= M) return;
    const float* xr = x + (long)row * Dd;
    float s = 0.f;
    for (int i = lane; i < Dd; i += 32) s += xr[i];
    for (int o = 16; o; o >>= 1) s += __shfl_xor(s, o, 32);
    float mean = s / (float)Dd;
    float v = 0.f;
    for (int i = lane; i < Dd; i += 32) { float d = xr[i] - mean; v += d * d; }
    for (int o = 16; o; o >>= 1) v += __shfl_xor(v, o, 32);
    float inv = rsqrtf(v / (float)Dd + 1e-5f);
    for (int i = lane; i < Dd; i += 32) {
        float y = (xr[i] - mean) * inv * g[i] + b[i];
        if (outB) outB[(long)row * Dd + i] = f2bf(y);
        if (outF) outF[(long)row * Dd + i] = y;
    }
}

// ---------------- row softmax: one wave per row, bf16 (padded ld) out -------
__launch_bounds__(256)
__global__ void k_softmax(const float* __restrict__ S, unsigned short* __restrict__ P,
                          int rows, int n, int ldp) {
    int row = blockIdx.x * 8 + (threadIdx.x >> 5);
    int lane = threadIdx.x & 31;
    if (row >= rows) return;
    const float* s = S + (long)row * n;
    unsigned short* p = P + (long)row * ldp;
    float mx = -3.4e38f;
    for (int i = lane; i < n; i += 32) mx = fmaxf(mx, s[i]);
    for (int o = 16; o; o >>= 1) mx = fmaxf(mx, __shfl_xor(mx, o, 32));
    float sum = 0.f;
    for (int i = lane; i < n; i += 32) sum += __expf(s[i] - mx);
    for (int o = 16; o; o >>= 1) sum += __shfl_xor(sum, o, 32);
    float inv = 1.f / sum;
    for (int i = lane; i < n; i += 32) p[i] = f2bf(__expf(s[i] - mx) * inv);
    for (int i = n + lane; i < ldp; i += 32) p[i] = 0;     // zero pad (K tail)
}

// ---------------- misc elementwise kernels ---------------------------------
__global__ void k_embed(const int* __restrict__ src, const float* __restrict__ emb,
                        float* __restrict__ xf, int BT, int T, int Dd) {
    long idx = (long)blockIdx.x * blockDim.x + threadIdx.x;
    if (idx >= (long)BT * Dd) return;
    int bt = (int)(idx / Dd), d = (int)(idx % Dd);
    int t = bt % T;
    xf[idx] = emb[(long)src[bt] * Dd + d] + posenc(t, d, Dd);
}

__global__ void k_cvt(const float* __restrict__ src, unsigned short* __restrict__ dst,
                      int Kd, int Nd, int trans) {
    long idx = (long)blockIdx.x * blockDim.x + threadIdx.x;
    if (idx >= (long)Kd * Nd) return;
    int k = (int)(idx / Nd), n = (int)(idx % Nd);
    float v = trans ? src[(long)n * Kd + k] : src[idx];
    dst[idx] = f2bf(v);
}

__global__ void k_csum(const int* __restrict__ dur, int* __restrict__ csum, int T) {
    if (threadIdx.x == 0 && blockIdx.x == 0) {
        int run = 0;
        for (int t = 0; t < T; ++t) { run += dur[t]; csum[t] = run; }
    }
}

// zero the per-batch pad rows [S, Spad) of the padded qkv tensor
__global__ void k_zpad(unsigned short* __restrict__ q, int S, int Spad, int ld, int Bn) {
    long idx = (long)blockIdx.x * blockDim.x + threadIdx.x;
    long per = (long)(Spad - S) * ld;
    if (idx >= per * Bn) return;
    int b = (int)(idx / per);
    long r = idx % per;
    q[(long)b * Spad * ld + (long)S * ld + r] = 0;
}

__global__ void k_expand(const float* __restrict__ xe, const int* __restrict__ csum,
                         const int* __restrict__ dur, float* __restrict__ xf,
                         int Bn, int T, int Tout, int Dd) {
    long idx = (long)blockIdx.x * blockDim.x + threadIdx.x;
    if (idx >= (long)Bn * Tout * Dd) return;
    int d = (int)(idx % Dd);
    long r = idx / Dd;
    int t = (int)(r % Tout);
    int b = (int)(r / Tout);
    int lo = 0, hi = T;                                   // searchsorted(right)
    while (lo < hi) { int mid = (lo + hi) >> 1; if (csum[mid] > t) hi = mid; else lo = mid + 1; }
    int tok = lo < T ? lo : T - 1;
    int start = csum[tok] - dur[tok];
    float alpha = (tok < T - 1) ? (float)(t - start) / (float)dur[tok] : 0.f;
    int nxt = (tok + 1 < T) ? tok + 1 : T - 1;
    float v = (1.f - alpha) * xe[((long)b * T + tok) * Dd + d] +
              alpha * xe[((long)b * T + nxt) * Dd + d];
    xf[idx] = v + posenc(t, d, Dd);
}

// ---------------------------------------------------------------------------
extern "C" void kernel_launch(void* const* d_in, const int* in_sizes, int n_in,
                              void* d_out, int out_size, void* d_ws, size_t ws_size,
                              hipStream_t stream) {
    (void)in_sizes; (void)n_in; (void)out_size; (void)ws_size;
    const int  Bn = 16, T = 256, Dd = 384, Hh = 2, HD = 192, FFd = 1536, Ll = 4, NM = 80;
    const int  TOUT = 2038;                                // sum of 4+(t%9), t<256
    const int  SPMAX = 2048;                               // padded token stride (dec)
    const int  MENC = Bn * T;                              // 4096
    const long MDEC = (long)Bn * TOUT;                     // 32608

    const int*   src  = (const int*)d_in[0];
    const int*   dur  = (const int*)d_in[1];
    const float* emb  = (const float*)d_in[2];
    // stacks: [ln1g ln1b wqkv bqkv wo bo ln2g ln2b w1 b1 w2 b2]
    const float* enc[12]; for (int i = 0; i < 12; ++i) enc[i] = (const float*)d_in[3 + i];
    const float* encNg = (const float*)d_in[15];
    const float* encNb = (const float*)d_in[16];
    const float* dec[12]; for (int i = 0; i < 12; ++i) dec[i] = (const float*)d_in[21 + i];
    const float* decNg = (const float*)d_in[33];
    const float* decNb = (const float*)d_in[34];
    const float* melW  = (const float*)d_in[35];
    const float* melB  = (const float*)d_in[36];

    // ---- workspace layout ----
    char* ws = (char*)d_ws;
    size_t off = 0;
    auto take = [&](size_t bytes) { size_t r = off; off += (bytes + 255) & ~(size_t)255; return r; };
    const long PERL = (long)Dd * 3 * Dd + (long)Dd * Dd + (long)Dd * FFd + (long)FFd * Dd;
    unsigned short* wbf   = (unsigned short*)(ws + take((size_t)(8 * PERL + (long)Dd * NM) * 2));
    int*            csum  = (int*)(ws + take(T * 4));
    float*          xf    = (float*)(ws + take((size_t)MDEC * Dd * 4));
    unsigned short* xb    = (unsigned short*)(ws + take((size_t)MDEC * Dd * 2));
    unsigned short* qkvb  = (unsigned short*)(ws + take((size_t)Bn * SPMAX * 3 * Dd * 2));
    unsigned short* ffb   = (unsigned short*)(ws + take((size_t)MDEC * FFd * 2));
    unsigned short* attno = (unsigned short*)(ws + take((size_t)MDEC * Dd * 2));
    float*          scor  = (float*)(ws + take((size_t)Hh * TOUT * TOUT * 4));
    unsigned short* pmat  = (unsigned short*)(ws + take((size_t)Hh * TOUT * SPMAX * 2));
    float*          xe    = (float*)(ws + take((size_t)MENC * Dd * 4));

    auto wlayer = [&](int stack, int l, int which) -> unsigned short* {
        unsigned short* base = wbf + (size_t)(stack * Ll + l) * PERL;
        if (which == 0) return base;                                   // qkvT (384x1152)
        if (which == 1) return base + (size_t)Dd * 3 * Dd;             // woT  (384x384)
        if (which == 2) return base + (size_t)Dd * 3 * Dd + (size_t)Dd * Dd;        // w1
        return base + (size_t)Dd * 3 * Dd + (size_t)Dd * Dd + (size_t)Dd * FFd;     // w2
    };
    unsigned short* melbf = wbf + (size_t)8 * PERL;

    auto cvt = [&](const float* s, unsigned short* d, int K, int N, int tr) {
        long n = (long)K * N;
        k_cvt<<<dim3((unsigned)((n + 255) / 256)), 256, 0, stream>>>(s, d, K, N, tr);
    };
    auto gemm = [&](const unsigned short* A, long sAz, int lda,
                    const unsigned short* Bm, long sBz, int ldb, int trB,
                    const float* bias, const float* resid,
                    float* oF, long sCFz, int ldcF,
                    unsigned short* oB, long sCBz, int ldcB,
                    long M, int N, int K, float scale, int relu, int trT, int nz) {
        dim3 g((unsigned)((M + 63) / 64), (unsigned)((N + 63) / 64), (unsigned)nz);
        k_gemm<<<g, 128, 0, stream>>>(A, sAz, lda, Bm, sBz, ldb, trB, bias, resid,
                                      oF, sCFz, ldcF, oB, sCBz, ldcB,
                                      (int)M, N, K, scale, relu, trT, TOUT);
    };
    auto ln = [&](const float* x, const float* g, const float* b,
                  unsigned short* oB, float* oF, long M) {
        k_ln<<<dim3((unsigned)((M + 7) / 8)), 256, 0, stream>>>(x, g, b, oB, oF, (int)M, Dd);
    };

    // ---- repack weights to bf16 (with transpose where torch does x@W.T) ----
    for (int s = 0; s < 2; ++s) {
        const float** p = s ? dec : enc;
        for (int l = 0; l < Ll; ++l) {
            cvt(p[2]  + (long)l * 3 * Dd * Dd, wlayer(s, l, 0), Dd, 3 * Dd, 1);
            cvt(p[4]  + (long)l * Dd * Dd,     wlayer(s, l, 1), Dd, Dd,     1);
            cvt(p[8]  + (long)l * Dd * FFd,    wlayer(s, l, 2), Dd, FFd,    0);
            cvt(p[10] + (long)l * FFd * Dd,    wlayer(s, l, 3), FFd, Dd,    0);
        }
    }
    cvt(melW, melbf, Dd, NM, 0);
    k_csum<<<1, 32, 0, stream>>>(dur, csum, T);

    // ---- embedding + positional encoding ----
    {
        long n = (long)MENC * Dd;
        k_embed<<<dim3((unsigned)((n + 255) / 256)), 256, 0, stream>>>(src, emb, xf, MENC, T, Dd);
    }

    float sscale = 1.0f / sqrtf((float)HD);
    auto run_stack = [&](const float** p, int stack, long M, int S, int Spad, int nB) {
        const int ldP  = Spad;                             // padded P row stride
        const long SP3 = (long)Spad * 3 * Dd;              // padded qkv batch stride
        for (int l = 0; l < Ll; ++l) {
            ln(xf, p[0] + l * Dd, p[1] + l * Dd, xb, nullptr, M);
            for (int b = 0; b < nB; ++b) {
                // qkv[b] = xln[b] @ WqkvT + b  -> bf16 (S x 1152, padded stride)
                gemm(xb + (size_t)b * S * Dd, 0, Dd, wlayer(stack, l, 0), 0, 3 * Dd, 0,
                     p[3] + l * 3 * Dd, nullptr, nullptr, 0, 0,
                     qkvb + (size_t)b * SP3, 0, 3 * Dd, S, 3 * Dd, Dd, 1.f, 0, 0, 1);
            }
            for (int b = 0; b < nB; ++b) {
                const unsigned short* qb = qkvb + (size_t)b * SP3;
                // scores[h] = Q Kt / sqrt(hd)   (f32, per batch)
                gemm(qb, HD, 3 * Dd, qb + Dd, HD, 3 * Dd, 1, nullptr, nullptr,
                     scor, (long)S * S, S, nullptr, 0, 0, S, S, HD, sscale, 0, 0, Hh);
                k_softmax<<<dim3((unsigned)((Hh * S + 7) / 8)), 256, 0, stream>>>(
                    scor, pmat, Hh * S, S, ldP);
                // O[h] = P V  (K padded to Spad; P and V pads are zero)
                gemm(pmat, (long)S * ldP, ldP, qb + 2 * Dd, HD, 3 * Dd, 0, nullptr, nullptr,
                     nullptr, 0, 0, attno + (size_t)b * S * Dd, HD, Dd,
                     S, HD, Spad, 1.f, 0, 0, Hh);
            }
            // h = h + O @ WoT + bo
            gemm(attno, 0, Dd, wlayer(stack, l, 1), 0, Dd, 0, p[5] + l * Dd, xf,
                 xf, 0, Dd, nullptr, 0, 0, M, Dd, Dd, 1.f, 0, 0, 1);
            ln(xf, p[6] + l * Dd, p[7] + l * Dd, xb, nullptr, M);
            gemm(xb, 0, Dd, wlayer(stack, l, 2), 0, FFd, 0, p[9] + l * FFd, nullptr,
                 nullptr, 0, 0, ffb, 0, FFd, M, FFd, Dd, 1.f, 1, 0, 1);
            gemm(ffb, 0, FFd, wlayer(stack, l, 3), 0, Dd, 0, p[11] + l * Dd, xf,
                 xf, 0, Dd, nullptr, 0, 0, M, Dd, FFd, 1.f, 0, 0, 1);
        }
    };

    run_stack(enc, 0, MENC, T, T /*Spad==S, no pads*/, Bn);
    ln(xf, encNg, encNb, nullptr, xe, MENC);               // encoder output (f32)

    {   // ragged expand + decoder positional encoding
        long n = (long)Bn * TOUT * Dd;
        k_expand<<<dim3((unsigned)((n + 255) / 256)), 256, 0, stream>>>(
            xe, csum, dur, xf, Bn, T, TOUT, Dd);
    }

    {   // zero decoder qkv pad rows (K-padding for the PV gemm)
        long n = (long)Bn * (SPMAX - TOUT) * 3 * Dd;
        k_zpad<<<dim3((unsigned)((n + 255) / 256)), 256, 0, stream>>>(
            qkvb, TOUT, SPMAX, 3 * Dd, Bn);
    }
    run_stack(dec, 1, MDEC, TOUT, SPMAX, Bn);
    ln(xf, decNg, decNb, xb, nullptr, MDEC);

    // mel = xln @ melW + melB, stored transposed to (B, n_mels, T')
    gemm(xb, 0, Dd, melbf, 0, NM, 0, melB, nullptr,
         (float*)d_out, 0, NM, nullptr, 0, 0, MDEC, NM, Dd, 1.f, 0, 1, 1);
}
